// GraphAtten_85804856640064
// MI455X (gfx1250) — matrix operand order repaired and checked by reading, not compiled, not run on hardware
//
#include <hip/hip_runtime.h>
#include <hip/hip_bf16.h>

// GraphAtten on MI455X (gfx1250, wave32, WMMA).
// Pipeline (all on `stream`, workspace in d_ws):
//   1. prep_w         : deinterleave + f16-convert nn_w -> Wh[256][512]
//   2. scores         : s1/s0 per-column dots of x with a_w halves (f32)
//   3. transpose_half : x[B,C,N] f32 -> ShT[B,N,0:256] f16 (LDS-tiled transpose)
//   4. attn_agg       : softmax over K=16 + weighted f16 gather-sum -> ShT[B,N,256:512]
//   5. gemm_wmma      : out = relu(Wh * ShT^T + b); per-batch 256x512x2048 GEMM,
//                       v_wmma_f32_16x16x32_f16, double-buffered LDS, b128-only
//                       LDS traffic (A and B tiles share one load pattern).

#define BATCH 8
#define CH    256
#define NPTS  2048
#define KNN   16
#define COUT  256
#define C2    512   // 2*CH

typedef _Float16 h8   __attribute__((ext_vector_type(8)));
typedef _Float16 v16h __attribute__((ext_vector_type(16)));
typedef float    v8f  __attribute__((ext_vector_type(8)));

// ---------------------------------------------------------------- kernel 1
// Wh[o][c2] : c2<256 -> nn_w[o][2*c2] (multiplies x), else nn_w[o][2*(c2-256)+1]
// (multiplies agg). Absorbs the torch interleaved channel stack into the weights.
__global__ __launch_bounds__(256) void prep_w(const float* __restrict__ nn_w,
                                              _Float16* __restrict__ Wh) {
  int i  = blockIdx.x * 256 + threadIdx.x;       // 0 .. 256*512-1
  int o  = i >> 9;
  int c2 = i & 511;
  int src = (c2 < CH) ? (2 * c2) : (2 * (c2 - CH) + 1);
  Wh[o * C2 + c2] = (_Float16)nn_w[o * C2 + src];
}

// ---------------------------------------------------------------- kernel 2
// Per (b,n): s1 = <x[:,n], a_w[:C]>, s0 = <x[:,n], a_w[C:]>.
// Threads = consecutive n -> every global access coalesced; x is L2-resident.
__global__ __launch_bounds__(256) void scores(const float* __restrict__ x,
                                              const float* __restrict__ a_w,
                                              float* __restrict__ s1,
                                              float* __restrict__ s0) {
  __shared__ float aw[C2];
  int t = threadIdx.x;
  aw[t]       = a_w[t];
  aw[t + 256] = a_w[t + 256];
  __syncthreads();
  int b = blockIdx.x >> 3;                       // 8 blocks per batch
  int n = ((blockIdx.x & 7) << 8) + t;
  const float* xb = x + (size_t)b * CH * NPTS;
  float acc1 = 0.f, acc0 = 0.f;
  for (int c = 0; c < CH; ++c) {
    float v = xb[(size_t)c * NPTS + n];
    acc1 += v * aw[c];
    acc0 += v * aw[CH + c];
  }
  s1[b * NPTS + n] = acc1;
  s0[b * NPTS + n] = acc0;
}

// ---------------------------------------------------------------- kernel 3
// LDS-tiled transpose + f16 convert: x[B,C,N] -> ShT[B][n][c] (c2 rows 0..255).
// Makes every neighbor gather a contiguous 512B f16 row, and lines the data up
// n-major for b128-only GEMM B-tile loads.
__global__ __launch_bounds__(256) void transpose_half(const float* __restrict__ x,
                                                      _Float16* __restrict__ ShT) {
  __shared__ float tile[32][33];
  int b  = blockIdx.z;
  int n0 = blockIdx.x * 32, c0 = blockIdx.y * 32;
  int tx = threadIdx.x & 31, ty = threadIdx.x >> 5;   // ty: 0..7
  const float* xb = x + (size_t)b * CH * NPTS;
  _Float16*   Sb = ShT + (size_t)b * NPTS * C2;
#pragma unroll
  for (int j = 0; j < 4; ++j)
    tile[ty + j * 8][tx] = xb[(size_t)(c0 + ty + j * 8) * NPTS + n0 + tx];
  __syncthreads();
#pragma unroll
  for (int j = 0; j < 4; ++j)
    Sb[(size_t)(n0 + ty + j * 8) * C2 + c0 + tx] = (_Float16)tile[tx][ty + j * 8];
}

// ---------------------------------------------------------------- kernel 4
// Block = (b, 16 consecutive n). Phase 1: 256 threads = 16 n x 16 k scores,
// softmax via 16-lane shuffles (softmax itself in f32). Phase 2: thread t =
// channel c, accumulate K=16 weighted gathers (f16 rows, L2-resident) for all
// 16 n; write ShT[n][256+c] -- fully coalesced (consecutive c = consecutive t).
__global__ __launch_bounds__(256) void attn_agg(const long long* __restrict__ ei,
                                                const float* __restrict__ s1,
                                                const float* __restrict__ s0,
                                                const float* __restrict__ a_b,
                                                _Float16* __restrict__ ShT) {
  __shared__ float wgt[16][16];
  __shared__ int   jj[16][16];
  int t  = threadIdx.x;
  int b  = blockIdx.x >> 7;                      // N/16 = 128 blocks per batch
  int n0 = (blockIdx.x & 127) << 4;
  {
    int nl = t >> 4, k = t & 15;
    size_t base = ((size_t)b * NPTS + (n0 + nl)) * KNN + k;
    int j0 = (int)ei[base];                                      // neighbor
    int j1 = (int)ei[(size_t)BATCH * NPTS * KNN + base];         // center
    float e = s1[b * NPTS + j1] + s0[b * NPTS + j0] + a_b[0];
    float m = e;
#pragma unroll
    for (int off = 8; off >= 1; off >>= 1) m = fmaxf(m, __shfl_xor(m, off, 16));
    float ex = __expf(e - m);
    float s = ex;
#pragma unroll
    for (int off = 8; off >= 1; off >>= 1) s += __shfl_xor(s, off, 16);
    wgt[nl][k] = ex / s;
    jj[nl][k]  = j0;
  }
  __syncthreads();
  _Float16* Sb = ShT + (size_t)b * NPTS * C2;
  float acc[16];
#pragma unroll
  for (int nl = 0; nl < 16; ++nl) acc[nl] = 0.f;
  for (int nl = 0; nl < 16; ++nl) {
#pragma unroll
    for (int k = 0; k < KNN; ++k)
      acc[nl] += wgt[nl][k] * (float)Sb[(size_t)jj[nl][k] * C2 + t];
  }
#pragma unroll
  for (int nl = 0; nl < 16; ++nl)
    Sb[(size_t)(n0 + nl) * C2 + CH + t] = (_Float16)acc[nl];
}

// ---------------------------------------------------------------- kernel 5
// Per-batch GEMM: out[o][n] = relu( sum_{c2<512} Wh[o][c2]*ShT[n][c2] + b[o] ).
// Both operands are [row][512] row-major -> identical b128 tile loaders.
// Block tile 128x128, BK=32, 8 waves; wave = 32x64 subtile = 2x4 f32 accums
// -> 8 v_wmma per k-step. Double-buffered LDS, one barrier per k-step.
#define BM 128
#define BN 128
#define BK 32
#define PK 40    // padded k-stride (halves) to break LDS bank conflicts
#define NSTEP (C2 / BK)

__global__ __launch_bounds__(256) void gemm_wmma(const _Float16* __restrict__ Wh,
                                                 const _Float16* __restrict__ ShT,
                                                 const float* __restrict__ nn_b,
                                                 float* __restrict__ out) {
  __shared__ _Float16 As[2][BM * PK];   // [m][k]
  __shared__ _Float16 Bs[2][BN * PK];   // [n][k]
  int b  = blockIdx.z;
  int n0 = blockIdx.x * BN;
  int m0 = blockIdx.y * BM;
  int t = threadIdx.x;
  int w = t >> 5, lane = t & 31;
  int g = lane >> 4, lr = lane & 15;
  int wm = w & 3;            // 4 m-slots of 32 rows
  int wn = w >> 2;           // 2 n-slots of 64 cols
  const _Float16* Sb = ShT + (size_t)b * NPTS * C2;

  v8f acc[2][4];
  const v8f vzero = {0.f, 0.f, 0.f, 0.f, 0.f, 0.f, 0.f, 0.f};
#pragma unroll
  for (int mt = 0; mt < 2; ++mt)
#pragma unroll
    for (int nt = 0; nt < 4; ++nt) acc[mt][nt] = vzero;

  // Tile loaders: 128 rows x 32 k per buffer; thread t owns (row t>>1, 16-half
  // chunk (t&1)*16). Identical pattern for A (Wh) and B (ShT).
  int ld_r = t >> 1, ld_ks = (t & 1) << 4;
  const _Float16* pA = Wh + (size_t)(m0 + ld_r) * C2 + ld_ks;
  const _Float16* pB = Sb + (size_t)(n0 + ld_r) * C2 + ld_ks;
  int ld_off = ld_r * PK + ld_ks;

  { // prologue: tile 0 -> buffer 0
    h8 a0 = *(const h8*)(pA),     a1 = *(const h8*)(pA + 8);
    h8 b0 = *(const h8*)(pB),     b1 = *(const h8*)(pB + 8);
    *(h8*)(&As[0][ld_off]) = a0;  *(h8*)(&As[0][ld_off + 8]) = a1;
    *(h8*)(&Bs[0][ld_off]) = b0;  *(h8*)(&Bs[0][ld_off + 8]) = b1;
  }
  __syncthreads();

  for (int s = 0; s < NSTEP; ++s) {
    int cur = s & 1;
    h8 a0, a1, b0, b1;
    if (s + 1 < NSTEP) {           // issue next-tile global loads early
      const _Float16* qA = pA + (s + 1) * BK;
      const _Float16* qB = pB + (s + 1) * BK;
      a0 = *(const h8*)(qA);  a1 = *(const h8*)(qA + 8);
      b0 = *(const h8*)(qB);  b1 = *(const h8*)(qB + 8);
      if (s + 2 < NSTEP)
        __builtin_prefetch(pB + (s + 2) * BK, 0, 0);   // global_prefetch_b8
    }

    // A fragment: lane (g,lr) -> row lr, K = {8g..8g+7, 16+8g..23+8g}
    v16h afr[2], bfr[4];
#pragma unroll
    for (int mt = 0; mt < 2; ++mt) {
      const _Float16* p = &As[cur][(wm * 32 + mt * 16 + lr) * PK];
      h8 c0 = *(const h8*)(p + 8 * g);
      h8 c1 = *(const h8*)(p + 16 + 8 * g);
      v16h a;
#pragma unroll
      for (int j = 0; j < 8; ++j) { a[j] = c0[j]; a[j + 8] = c1[j]; }
      afr[mt] = a;
    }
    // B fragment: lane (g,lr) -> col lr, K = {8g..8g+7, 16+8g..23+8g}
#pragma unroll
    for (int nt = 0; nt < 4; ++nt) {
      const _Float16* p = &Bs[cur][(wn * 64 + nt * 16 + lr) * PK + 8 * g];
      h8 c0 = *(const h8*)(p);
      h8 c1 = *(const h8*)(p + 16);
      v16h bb;
#pragma unroll
      for (int j = 0; j < 8; ++j) { bb[j] = c0[j]; bb[j + 8] = c1[j]; }
      bfr[nt] = bb;
    }
#pragma unroll
    for (int mt = 0; mt < 2; ++mt)
#pragma unroll
      for (int nt = 0; nt < 4; ++nt)
        acc[mt][nt] = __builtin_amdgcn_wmma_f32_16x16x32_f16(
            false, afr[mt], false, bfr[nt], (short)0, acc[mt][nt], false, false);

    if (s + 1 < NSTEP) {           // park next tile in the other buffer
      int nxt = cur ^ 1;
      *(h8*)(&As[nxt][ld_off]) = a0;  *(h8*)(&As[nxt][ld_off + 8]) = a1;
      *(h8*)(&Bs[nxt][ld_off]) = b0;  *(h8*)(&Bs[nxt][ld_off + 8]) = b1;
    }
    __syncthreads();
  }

  // Epilogue: D layout -> lane (g,lr): row m = base + i + 8g, col n = lr.
  float* outb = out + (size_t)b * COUT * NPTS;
#pragma unroll
  for (int mt = 0; mt < 2; ++mt) {
    int mbase = m0 + wm * 32 + mt * 16 + 8 * g;
#pragma unroll
    for (int nt = 0; nt < 4; ++nt) {
      int n = n0 + wn * 64 + nt * 16 + lr;
#pragma unroll
      for (int i = 0; i < 8; ++i) {
        int m = mbase + i;
        float v = acc[mt][nt][i] + nn_b[m];
        outb[(size_t)m * NPTS + n] = fmaxf(v, 0.f);
      }
    }
  }
}

// ---------------------------------------------------------------- launcher
extern "C" void kernel_launch(void* const* d_in, const int* in_sizes, int n_in,
                              void* d_out, int out_size, void* d_ws, size_t ws_size,
                              hipStream_t stream) {
  const float*     x    = (const float*)d_in[0];
  const long long* ei   = (const long long*)d_in[1];   // int64 edge_index [2,B,N,K]
  const float*     a_w  = (const float*)d_in[2];
  const float*     a_b  = (const float*)d_in[3];
  const float*     nn_w = (const float*)d_in[4];
  const float*     nn_b = (const float*)d_in[5];
  float*           out  = (float*)d_out;

  char* ws = (char*)d_ws;
  size_t off = 0;
  _Float16* ShT = (_Float16*)(ws + off); off += (size_t)BATCH * NPTS * C2 * 2;  // 16 MB
  float* s1 = (float*)(ws + off);        off += (size_t)BATCH * NPTS * 4;
  float* s0 = (float*)(ws + off);        off += (size_t)BATCH * NPTS * 4;
  _Float16* Wh = (_Float16*)(ws + off);  off += (size_t)COUT * C2 * 2;

  prep_w<<<(COUT * C2) / 256, 256, 0, stream>>>(nn_w, Wh);
  scores<<<(BATCH * NPTS) / 256, 256, 0, stream>>>(x, a_w, s1, s0);
  transpose_half<<<dim3(NPTS / 32, CH / 32, BATCH), 256, 0, stream>>>(x, ShT);
  attn_agg<<<(BATCH * NPTS) / KNN, 256, 0, stream>>>(ei, s1, s0, a_b, ShT);
  gemm_wmma<<<dim3(NPTS / BN, COUT / BM, BATCH), 256, 0, stream>>>(Wh, ShT, nn_b, out);
}
